// LSTMCNNCRF_36661840839809
// MI455X (gfx1250) — matrix operand order repaired and checked by reading, not compiled, not run on hardware
//
#include <hip/hip_runtime.h>
#include <math.h>

typedef __attribute__((ext_vector_type(16))) __bf16 v16bf;
typedef __attribute__((ext_vector_type(8)))  float  v8f;

// ---------- helpers ----------
__device__ __forceinline__ unsigned short f2bfbits(float x) {
  union { float f; unsigned u; } v; v.f = x;
  unsigned r = v.u + 0x7FFFu + ((v.u >> 16) & 1u);
  return (unsigned short)(r >> 16);
}

// async copy of 16 bytes global -> LDS per lane (ASYNCcnt-tracked, gfx1250)
__device__ __forceinline__ void async_load_b128(unsigned lds_off, unsigned goff,
                                                unsigned long long sbase) {
  asm volatile("global_load_async_to_lds_b128 %0, %1, %2"
               :: "v"(lds_off), "v"(goff), "s"(sbase)
               : "memory");
}
__device__ __forceinline__ void wait_async_le2() {
  asm volatile("s_wait_asynccnt 0x2" ::: "memory");
}
__device__ __forceinline__ void wait_async_0() {
  asm volatile("s_wait_asynccnt 0x0" ::: "memory");
}

__device__ __forceinline__ bool allowedT(int i, int j) {
  if (i == 0 || i == 100) return (j == 0) || (j == 1);
  if (i == 99) return (j == 99) || (j == 100);
  return (j == i + 1) || (i >= 4 && j == 100); // i in 1..98
}

// ---------- 0) (B,D,L) -> (B*L, D) transpose + positional encoding, emit bf16 ----------
__global__ __launch_bounds__(256) void posenc_transpose(const float* __restrict__ emb,
                                                        unsigned short* __restrict__ out) {
  __shared__ float tile[32][33];
  const int b  = blockIdx.z;
  const int d0 = blockIdx.y << 5;
  const int l0 = blockIdx.x << 5;
  const int tx = threadIdx.x & 31, ty = threadIdx.x >> 5; // ty 0..7
#pragma unroll
  for (int rr = 0; rr < 4; rr++) {
    int d = d0 + ty + (rr << 3);
    tile[ty + (rr << 3)][tx] = emb[((size_t)b * 1536 + d) * 1024 + (l0 + tx)];
  }
  __syncthreads();
#pragma unroll
  for (int rr = 0; rr < 4; rr++) {
    int l = l0 + ty + (rr << 3);
    int d = d0 + tx;
    float p2   = (float)(d & ~1);
    float freq = __expf(p2 * (-9.210340371976184f / 1536.f));
    float ang  = (float)l * freq;
    float pe   = (d & 1) ? __cosf(ang) : __sinf(ang);
    out[((size_t)b * 1024 + l) * 1536 + d] = f2bfbits(tile[tx][ty + (rr << 3)] + pe);
  }
}

// ---------- 1) weight prep: f32 -> bf16 in GEMM-tiled [nblk][kblk][n64][k32] layout ----------
// doT: in is (N,K) and we want B[k][n] = in[n][k]
__global__ void prep_b_tiled(const float* __restrict__ in, unsigned short* __restrict__ out,
                             int K, int N, int doT) {
  int i = blockIdx.x * 256 + threadIdx.x;
  if (i >= K * N) return;
  int k = i / N, n = i - k * N;
  float v = doT ? in[n * K + k] : in[i];
  int nblk = n >> 6, nl = n & 63;
  int kblk = k >> 5, kl = k & 31;
  int nk = K >> 5;
  out[(size_t)((nblk * nk + kblk) * 2048 + nl * 32 + kl)] = f2bfbits(v);
}

// ---------- 2) WMMA bf16 GEMM with double-buffered async-to-LDS staging ----------
// C[M,N](f32) = A[M,K](bf16, row-major) * Bt(bf16, tiled) + bias1 + bias2
// block = 256 threads = 8 waves, tile 64x64, k-chunk 32, LDS row stride 40 halves (80B)
#define AST 40
__global__ __launch_bounds__(256) void gemm_wmma(const unsigned short* __restrict__ A,
                                                 const unsigned short* __restrict__ Bt,
                                                 const float* __restrict__ bias1,
                                                 const float* __restrict__ bias2,
                                                 float* __restrict__ C,
                                                 int M, int N, int K) {
  __shared__ __align__(16) unsigned short a_sm[2][64 * AST];
  __shared__ __align__(16) unsigned short b_sm[2][64 * AST];
  const int tid  = threadIdx.x;
  const int wave = tid >> 5, lane = tid & 31;
  const int hi = lane >> 4, lm = lane & 15;
  const int mt = wave & 3;       // 4 M-tiles of 16
  const int ng = wave >> 2;      // 2 N-groups of 32
  const int m0 = blockIdx.x * 64, n0 = blockIdx.y * 64;
  const int nk = K >> 5;
  const int nblk = blockIdx.y;

  // one 16B async copy per thread per matrix per k-chunk
  const int ar = tid >> 2, ac = tid & 3; // row 0..63, 16B chunk 0..3
  const unsigned a_lds[2] = {
    (unsigned)(size_t)&a_sm[0][ar * AST] + (unsigned)(ac * 16),
    (unsigned)(size_t)&a_sm[1][ar * AST] + (unsigned)(ac * 16) };
  const unsigned b_lds[2] = {
    (unsigned)(size_t)&b_sm[0][ar * AST] + (unsigned)(ac * 16),
    (unsigned)(size_t)&b_sm[1][ar * AST] + (unsigned)(ac * 16) };
  const unsigned long long abase = (unsigned long long)(size_t)A;
  const unsigned long long bbase = (unsigned long long)(size_t)Bt;
  const unsigned arow = (unsigned)(m0 + ar) * (unsigned)K;

  auto issue = [&](int ki, int buf) {
    unsigned aoff = (arow + ((unsigned)ki << 5)) * 2u + (unsigned)(ac * 16);
    async_load_b128(a_lds[buf], aoff, abase);
    unsigned boff = (unsigned)((nblk * nk + ki) * 2048 + ar * 32) * 2u + (unsigned)(ac * 16);
    async_load_b128(b_lds[buf], boff, bbase);
  };

  issue(0, 0);
  v8f acc0 = {}, acc1 = {};
  for (int ki = 0; ki < nk; ki++) {
    const int cur = ki & 1;
    if (ki + 1 < nk) { issue(ki + 1, cur ^ 1); wait_async_le2(); }
    else             { wait_async_0(); }
    __syncthreads();  // staged chunk visible to whole block
    const unsigned short* as = a_sm[cur];
    const unsigned short* bs = b_sm[cur];
    v16bf af, bf0, bf1;
    unsigned* au  = (unsigned*)&af;
    unsigned* b0u = (unsigned*)&bf0;
    unsigned* b1u = (unsigned*)&bf1;
#pragma unroll
    for (int q = 0; q < 8; q++) {
      int ka = ((q >> 2) << 4) + (hi << 3) + ((q & 3) << 1);
      au[q] = *(const unsigned*)&as[(mt * 16 + lm) * AST + ka];
      int kb = ((q >> 2) << 4) + (hi << 3) + ((q & 3) << 1);
      b0u[q] = *(const unsigned*)&bs[(ng * 32 + lm) * AST + kb];
      b1u[q] = *(const unsigned*)&bs[(ng * 32 + 16 + lm) * AST + kb];
    }
    acc0 = __builtin_amdgcn_wmma_f32_16x16x32_bf16(false, af, false, bf0, (short)0, acc0, false, false);
    acc1 = __builtin_amdgcn_wmma_f32_16x16x32_bf16(false, af, false, bf1, (short)0, acc1, false, false);
    __syncthreads();  // all reads done before this buffer is refilled
  }
  const int n_a = n0 + ng * 32 + lm;
  const int n_b = n_a + 16;
  const float ba = (bias1 ? bias1[n_a] : 0.f) + (bias2 ? bias2[n_a] : 0.f);
  const float bb = (bias1 ? bias1[n_b] : 0.f) + (bias2 ? bias2[n_b] : 0.f);
#pragma unroll
  for (int r = 0; r < 8; r++) {
    int m = m0 + mt * 16 + r + (hi << 3);
    C[(size_t)m * N + n_a] = acc0[r] + ba;
    C[(size_t)m * N + n_b] = acc1[r] + bb;
  }
}

// ---------- 3) LayerNorm(C=256) + ReLU, in place ----------
__global__ __launch_bounds__(256) void ln_relu(float* __restrict__ buf,
                                               const float* __restrict__ g,
                                               const float* __restrict__ b) {
  __shared__ float red[256];
  const int row = blockIdx.x, tid = threadIdx.x;
  float x = buf[(size_t)row * 256 + tid];
  red[tid] = x; __syncthreads();
  for (int s = 128; s > 0; s >>= 1) { if (tid < s) red[tid] += red[tid + s]; __syncthreads(); }
  float mean = red[0] / 256.f; __syncthreads();
  float d = x - mean;
  red[tid] = d * d; __syncthreads();
  for (int s = 128; s > 0; s >>= 1) { if (tid < s) red[tid] += red[tid + s]; __syncthreads(); }
  float var = red[0] / 256.f;
  float y = d * rsqrtf(var + 1e-5f) * g[tid] + b[tid];
  buf[(size_t)row * 256 + tid] = fmaxf(y, 0.f);
}

// ---------- 4) conv1d k=3, 256->64, same-pad, ReLU; bf16 out (feeds WMMA GEMMs) ----------
__global__ __launch_bounds__(256) void conv1_relu(const float* __restrict__ x,
                                                  const float* __restrict__ w,
                                                  const float* __restrict__ bias,
                                                  unsigned short* __restrict__ out) {
  const int f = threadIdx.x & 63, li = threadIdx.x >> 6;
  const int blk = blockIdx.x;         // 32 * 256
  const int b = blk >> 8, lg = blk & 255;
  const int l = (lg << 2) + li;
  float acc = bias[f];
#pragma unroll
  for (int dk = 0; dk < 3; dk++) {
    int ls = l + dk - 1;
    if (ls < 0 || ls >= 1024) continue;
    const float* xr = x + (((size_t)(b << 10) + ls) << 8);
    __builtin_prefetch(xr + 256, 0, 0); // next row -> global_prefetch_b8
    const float* wr = w + (size_t)f * 768 + dk;  // (64,256,3)
    for (int c = 0; c < 256; c++) acc += xr[c] * wr[c * 3];
  }
  out[(((size_t)(b << 10) + l) << 6) + f] = f2bfbits(fmaxf(acc, 0.f));
}

// ---------- 5) BiLSTM recurrence, WMMA-based: one persistent WG per direction ----------
// gates(32x512) = h(32x128) @ Whh^T ; Whh kept resident in registers as B-fragments.
#define LGS 516
__global__ __launch_bounds__(1024) void lstm_wmma(const float* __restrict__ xW_f,
                                                  const float* __restrict__ xW_b,
                                                  const float* __restrict__ Whh_f,
                                                  const float* __restrict__ Whh_b,
                                                  float* __restrict__ hout) {
  extern __shared__ char smem[];
  unsigned short* hlds = (unsigned short*)smem;              // 32 x 130 bf16
  float* gates = (float*)(smem + ((32 * 130 * 2 + 255) & ~255)); // 32 x LGS f32
  const int dir = blockIdx.x;
  const float* xw = dir ? xW_b : xW_f;
  const float* W  = dir ? Whh_b : Whh_f;                     // (512,128)
  const int tid = threadIdx.x;
  const int wave = tid >> 5, lane = tid & 31, hi = lane >> 4, lm = lane & 15;
  const int mt = wave & 1;                                   // 2 M-tiles (batch 0-15 / 16-31)
  const int np = wave >> 1;                                  // 16 N-pair groups
  // resident B fragments: B[k][n] = Whh[n][k]
  v16bf bfr[2][4];
#pragma unroll
  for (int tt = 0; tt < 2; tt++) {
    const int n = (np * 2 + tt) * 16 + lm;
    const float* wr = W + (size_t)n * 128;
#pragma unroll
    for (int kc = 0; kc < 4; kc++) {
      unsigned* bu = (unsigned*)&bfr[tt][kc];
#pragma unroll
      for (int q = 0; q < 8; q++) {
        int k = kc * 32 + ((q >> 2) << 4) + (hi << 3) + ((q & 3) << 1);
        unsigned lo  = f2bfbits(wr[k]);
        unsigned hi2 = f2bfbits(wr[k + 1]);
        bu[q] = lo | (hi2 << 16);
      }
    }
  }
  for (int i = tid; i < 32 * 130; i += 1024) hlds[i] = 0;
  float cst[4] = {0.f, 0.f, 0.f, 0.f};
  __syncthreads();
  const int na = (np * 2) * 16 + lm, nb = na + 16;
  for (int s = 0; s < 1024; s++) {
    const int t = dir ? (1023 - s) : s;
    v8f acc0 = {}, acc1 = {};
#pragma unroll
    for (int kc = 0; kc < 4; kc++) {
      v16bf af;
      unsigned* au = (unsigned*)&af;
#pragma unroll
      for (int q = 0; q < 8; q++) {
        int k = kc * 32 + ((q >> 2) << 4) + (hi << 3) + ((q & 3) << 1);
        au[q] = *(const unsigned*)&hlds[(mt * 16 + lm) * 130 + k];
      }
      acc0 = __builtin_amdgcn_wmma_f32_16x16x32_bf16(false, af, false, bfr[0][kc], (short)0, acc0, false, false);
      acc1 = __builtin_amdgcn_wmma_f32_16x16x32_bf16(false, af, false, bfr[1][kc], (short)0, acc1, false, false);
    }
#pragma unroll
    for (int r = 0; r < 8; r++) {
      int m = mt * 16 + r + (hi << 3);
      gates[m * LGS + na] = acc0[r];
      gates[m * LGS + nb] = acc1[r];
    }
    __syncthreads();
    // elementwise cell update: 4 hidden units per thread
#pragma unroll
    for (int r = 0; r < 4; r++) {
      int idx = tid + (r << 10);
      int b = idx >> 7, j = idx & 127;
      const float* xr = xw + (((size_t)(b << 10) + t) << 9);
      const float* gr = gates + b * LGS;
      float gi = gr[j]       + xr[j];
      float gf = gr[128 + j] + xr[128 + j];
      float gg = gr[256 + j] + xr[256 + j];
      float go = gr[384 + j] + xr[384 + j];
      float si = 1.f / (1.f + __expf(-gi));
      float sf = 1.f / (1.f + __expf(-gf));
      float so = 1.f / (1.f + __expf(-go));
      float tg = tanhf(gg);
      cst[r] = sf * cst[r] + si * tg;
      float hv = so * tanhf(cst[r]);
      hlds[b * 130 + j] = f2bfbits(hv);
      hout[(((size_t)(b << 10) + t) << 8) + (dir ? 128 : 0) + j] = hv;
    }
    __syncthreads();
  }
}

// ---------- 6) conv1d k=5, 256->128, same-pad, ReLU ----------
__global__ __launch_bounds__(256) void conv2_relu(const float* __restrict__ x,
                                                  const float* __restrict__ w,
                                                  const float* __restrict__ bias,
                                                  float* __restrict__ out) {
  const int f = threadIdx.x & 127, li = threadIdx.x >> 7;
  const int blk = blockIdx.x;          // 32 * 512
  const int b = blk >> 9, lg = blk & 511;
  const int l = (lg << 1) + li;
  float acc = bias[f];
#pragma unroll
  for (int dk = 0; dk < 5; dk++) {
    int ls = l + dk - 2;
    if (ls < 0 || ls >= 1024) continue;
    const float* xr = x + (((size_t)(b << 10) + ls) << 8);
    __builtin_prefetch(xr + 256, 0, 0);
    const float* wr = w + (size_t)f * 1280 + dk;  // (128,256,5)
    for (int c = 0; c < 256; c++) acc += xr[c] * wr[c * 5];
  }
  out[(((size_t)(b << 10) + l) << 7) + f] = fmaxf(acc, 0.f);
}

// ---------- 7) emissions: feats(.,128)@W_emis(128,2)+b, expand to 101 states ----------
__global__ __launch_bounds__(128) void emis_kernel(const float* __restrict__ feats,
                                                   const float* __restrict__ We,
                                                   const float* __restrict__ be,
                                                   float* __restrict__ emis) {
  __shared__ float r0[128], r1[128];
  __shared__ float e01[2];
  const int row = blockIdx.x, tid = threadIdx.x;
  float x = feats[(size_t)row * 128 + tid];
  r0[tid] = x * We[tid * 2];
  r1[tid] = x * We[tid * 2 + 1];
  __syncthreads();
  for (int s = 64; s > 0; s >>= 1) {
    if (tid < s) { r0[tid] += r0[tid + s]; r1[tid] += r1[tid + s]; }
    __syncthreads();
  }
  if (tid == 0) { e01[0] = r0[0] + be[0]; e01[1] = r1[0] + be[1]; }
  __syncthreads();
  if (tid < 101) emis[(size_t)row * 101 + tid] = (tid == 0) ? e01[0] : e01[1];
}

// ---------- 8) CRF forward (alphas + logZ), one WG per batch ----------
__global__ __launch_bounds__(128) void crf_forward(const float* __restrict__ emis,
                                                   const float* __restrict__ trans,
                                                   const float* __restrict__ startv,
                                                   const float* __restrict__ endv,
                                                   float* __restrict__ alphas,
                                                   float* __restrict__ logZ) {
  __shared__ float tr[101 * 102];
  __shared__ float av[104];
  const int b = blockIdx.x, tid = threadIdx.x;
  for (int i = tid; i < 101 * 101; i += 128) {
    int r = i / 101, c = i - r * 101;
    tr[r * 102 + c] = allowedT(r, c) ? trans[i] : -1e4f;
  }
  const float* e = emis + (size_t)b * 1024 * 101;
  float* al = alphas + (size_t)b * 1024 * 101;
  __syncthreads();
  if (tid < 101) { float a = startv[tid] + e[tid]; av[tid] = a; al[tid] = a; }
  __syncthreads();
  for (int t = 1; t < 1024; t++) {
    float vnew = 0.f;
    if (tid < 101) {
      float m = -3.4e38f, s2 = 0.f;
      for (int i = 0; i < 101; i++) {
        float v = av[i] + tr[i * 102 + tid];
        if (v > m) { s2 = s2 * __expf(m - v) + 1.f; m = v; }
        else       { s2 += __expf(v - m); }
      }
      vnew = m + __logf(s2) + e[(size_t)t * 101 + tid];
    }
    __syncthreads();
    if (tid < 101) { av[tid] = vnew; al[(size_t)t * 101 + tid] = vnew; }
    __syncthreads();
  }
  if (tid == 0) {
    float m = -3.4e38f, s2 = 0.f;
    for (int j = 0; j < 101; j++) {
      float ev = (j == 0 || j == 100) ? endv[j] : -1e4f;
      float v = av[j] + ev;
      if (v > m) { s2 = s2 * __expf(m - v) + 1.f; m = v; } else { s2 += __expf(v - m); }
    }
    logZ[b] = m + __logf(s2);
  }
}

// ---------- 9) CRF backward (betas) ----------
__global__ __launch_bounds__(128) void crf_backward(const float* __restrict__ emis,
                                                    const float* __restrict__ trans,
                                                    const float* __restrict__ endv,
                                                    float* __restrict__ betas) {
  __shared__ float tr[101 * 102];
  __shared__ float bv[104];
  const int b = blockIdx.x, tid = threadIdx.x;
  for (int i = tid; i < 101 * 101; i += 128) {
    int r = i / 101, c = i - r * 101;
    tr[r * 102 + c] = allowedT(r, c) ? trans[i] : -1e4f;
  }
  const float* e = emis + (size_t)b * 1024 * 101;
  float* be = betas + (size_t)b * 1024 * 101;
  __syncthreads();
  if (tid < 101) {
    float v = (tid == 0 || tid == 100) ? endv[tid] : -1e4f;
    bv[tid] = v;
    be[(size_t)1023 * 101 + tid] = v;
  }
  __syncthreads();
  for (int t = 1022; t >= 0; t--) {
    float vnew = 0.f;
    if (tid < 101) {
      float m = -3.4e38f, s2 = 0.f;
      const float* et = e + (size_t)(t + 1) * 101;
      for (int j = 0; j < 101; j++) {
        float v = tr[tid * 102 + j] + et[j] + bv[j];
        if (v > m) { s2 = s2 * __expf(m - v) + 1.f; m = v; }
        else       { s2 += __expf(v - m); }
      }
      vnew = m + __logf(s2);
    }
    __syncthreads();
    if (tid < 101) { bv[tid] = vnew; be[(size_t)t * 101 + tid] = vnew; }
    __syncthreads();
  }
}

// ---------- 10) marginals: softmax(alpha+beta) per (b,l) ----------
__global__ __launch_bounds__(128) void marginals(const float* __restrict__ alphas,
                                                 const float* __restrict__ betas,
                                                 float* __restrict__ probs) {
  __shared__ float red[128];
  const int row = blockIdx.x, tid = threadIdx.x;
  float v = -3.4e38f;
  if (tid < 101) v = alphas[(size_t)row * 101 + tid] + betas[(size_t)row * 101 + tid];
  red[tid] = v; __syncthreads();
  for (int s = 64; s > 0; s >>= 1) { if (tid < s) red[tid] = fmaxf(red[tid], red[tid + s]); __syncthreads(); }
  float m = red[0]; __syncthreads();
  float ex = (tid < 101) ? __expf(v - m) : 0.f;
  red[tid] = ex; __syncthreads();
  for (int s = 64; s > 0; s >>= 1) { if (tid < s) red[tid] += red[tid + s]; __syncthreads(); }
  float sum = red[0];
  if (tid < 101) probs[(size_t)row * 101 + tid] = ex / sum;
}

// ---------- 11) Viterbi + backtrack ----------
__global__ __launch_bounds__(128) void crf_viterbi(const float* __restrict__ emis,
                                                   const float* __restrict__ trans,
                                                   const float* __restrict__ startv,
                                                   const float* __restrict__ endv,
                                                   const float* __restrict__ logZ,
                                                   unsigned char* __restrict__ ptrs,
                                                   float* __restrict__ paths,
                                                   float* __restrict__ pprobs) {
  __shared__ float tr[101 * 102];
  __shared__ float vv[104];
  const int b = blockIdx.x, tid = threadIdx.x;
  for (int i = tid; i < 101 * 101; i += 128) {
    int r = i / 101, c = i - r * 101;
    tr[r * 102 + c] = allowedT(r, c) ? trans[i] : -1e4f;
  }
  const float* e = emis + (size_t)b * 1024 * 101;
  unsigned char* pb = ptrs + (size_t)b * 1024 * 101;
  __syncthreads();
  if (tid < 101) vv[tid] = startv[tid] + e[tid];
  __syncthreads();
  for (int t = 1; t < 1024; t++) {
    float vnew = 0.f; int arg = 0;
    if (tid < 101) {
      float best = -3.4e38f;
      for (int i = 0; i < 101; i++) {
        float sc = vv[i] + tr[i * 102 + tid];
        if (sc > best) { best = sc; arg = i; }
      }
      vnew = best + e[(size_t)t * 101 + tid];
      pb[(size_t)t * 101 + tid] = (unsigned char)arg;
    }
    __syncthreads();
    if (tid < 101) vv[tid] = vnew;
    __syncthreads();
  }
  if (tid == 0) {
    float best = -3.4e38f; int last = 0;
    for (int j = 0; j < 101; j++) {
      float f = vv[j] + ((j == 0 || j == 100) ? endv[j] : -1e4f);
      if (f > best) { best = f; last = j; }
    }
    pprobs[b] = __expf(best - logZ[b]);
    int st = last;
    paths[(size_t)b * 1024 + 1023] = (float)st;
    for (int t = 1023; t >= 1; t--) {
      st = pb[(size_t)t * 101 + st];
      paths[(size_t)b * 1024 + (t - 1)] = (float)st;
    }
  }
}

// ---------- launch ----------
extern "C" void kernel_launch(void* const* d_in, const int* in_sizes, int n_in,
                              void* d_out, int out_size, void* d_ws, size_t ws_size,
                              hipStream_t stream) {
  (void)in_sizes; (void)n_in; (void)out_size; (void)ws_size;
  const float* emb  = (const float*)d_in[0];
  const float* Wbn  = (const float*)d_in[3];
  const float* bbn  = (const float*)d_in[4];
  const float* lng  = (const float*)d_in[5];
  const float* lnb  = (const float*)d_in[6];
  const float* c1w  = (const float*)d_in[7];
  const float* c1b  = (const float*)d_in[8];
  const float* Wihf = (const float*)d_in[9];
  const float* Whhf = (const float*)d_in[10];
  const float* bihf = (const float*)d_in[11];
  const float* bhhf = (const float*)d_in[12];
  const float* Wihb = (const float*)d_in[13];
  const float* Whhb = (const float*)d_in[14];
  const float* bihb = (const float*)d_in[15];
  const float* bhhb = (const float*)d_in[16];
  const float* c2w  = (const float*)d_in[17];
  const float* c2b  = (const float*)d_in[18];
  const float* Wem  = (const float*)d_in[19];
  const float* bem  = (const float*)d_in[20];
  const float* ctr  = (const float*)d_in[21];
  const float* cst  = (const float*)d_in[22];
  const float* cen  = (const float*)d_in[23];

  char* ws = (char*)d_ws;
  size_t off = 0;
  auto alloc = [&](size_t bytes) { size_t o = off; off += (bytes + 255) & ~(size_t)255; return o; };
  const size_t BL = 32 * 1024;
  unsigned short* Ape   = (unsigned short*)(ws + alloc(BL * 1536 * 2));
  unsigned short* WbBf  = (unsigned short*)(ws + alloc(1536 * 256 * 2));
  unsigned short* WifBf = (unsigned short*)(ws + alloc(64 * 512 * 2));
  unsigned short* WibBf = (unsigned short*)(ws + alloc(64 * 512 * 2));
  float* bneck          = (float*)(ws + alloc(BL * 256 * 4));
  unsigned short* c1o   = (unsigned short*)(ws + alloc(BL * 64 * 2));
  float* xWf            = (float*)(ws + alloc(BL * 512 * 4));
  float* xWb            = (float*)(ws + alloc(BL * 512 * 4));
  float* hcat           = (float*)(ws + alloc(BL * 256 * 4));
  float* c2o            = (float*)(ws + alloc(BL * 128 * 4));
  float* emis           = (float*)(ws + alloc(BL * 101 * 4));
  float* alphas         = (float*)(ws + alloc(BL * 101 * 4));
  float* betas          = (float*)(ws + alloc(BL * 101 * 4));
  float* logZ           = (float*)(ws + alloc(32 * 4));
  unsigned char* ptrs   = (unsigned char*)(ws + alloc(BL * 101));

  float* probs_out = (float*)d_out;
  float* paths_out = probs_out + (size_t)BL * 101;
  float* pprob_out = paths_out + BL;

  posenc_transpose<<<dim3(32, 48, 32), 256, 0, stream>>>(emb, Ape);
  prep_b_tiled<<<(1536 * 256 + 255) / 256, 256, 0, stream>>>(Wbn, WbBf, 1536, 256, 0);
  prep_b_tiled<<<(64 * 512 + 255) / 256, 256, 0, stream>>>(Wihf, WifBf, 64, 512, 1);
  prep_b_tiled<<<(64 * 512 + 255) / 256, 256, 0, stream>>>(Wihb, WibBf, 64, 512, 1);

  gemm_wmma<<<dim3(512, 4), 256, 0, stream>>>(Ape, WbBf, bbn, nullptr, bneck, 32768, 256, 1536);
  ln_relu<<<32768, 256, 0, stream>>>(bneck, lng, lnb);
  conv1_relu<<<32 * 256, 256, 0, stream>>>(bneck, c1w, c1b, c1o);

  gemm_wmma<<<dim3(512, 8), 256, 0, stream>>>(c1o, WifBf, bihf, bhhf, xWf, 32768, 512, 64);
  gemm_wmma<<<dim3(512, 8), 256, 0, stream>>>(c1o, WibBf, bihb, bhhb, xWb, 32768, 512, 64);

  size_t lstm_lds = (size_t)((32 * 130 * 2 + 255) & ~255) + (size_t)32 * LGS * 4;
  lstm_wmma<<<2, 1024, lstm_lds, stream>>>(xWf, xWb, Whhf, Whhb, hcat);

  conv2_relu<<<32 * 512, 256, 0, stream>>>(hcat, c2w, c2b, c2o);
  emis_kernel<<<32768, 128, 0, stream>>>(c2o, Wem, bem, emis);

  crf_forward<<<32, 128, 0, stream>>>(emis, ctr, cst, cen, alphas, logZ);
  crf_backward<<<32, 128, 0, stream>>>(emis, ctr, cen, betas);
  marginals<<<32768, 128, 0, stream>>>(alphas, betas, probs_out);
  crf_viterbi<<<32, 128, 0, stream>>>(emis, ctr, cst, cen, logZ, ptrs, paths_out, pprob_out);
}
